// SpatPPI_82970178224284
// MI455X (gfx1250) — compile-verified, matrix operand order checked by reading, not run on hardware
//
#include <hip/hip_runtime.h>

// SpatPPI neighborhood attention, fully fused: one workgroup (8 wave32s) per node.
// B=2, N=2000, K=32 neighbors, H=128, 4 heads x d=32.
// Pre-pass converts W_K1/W_K2/W_V to bf16 in d_ws (L2-resident, 128 KB).
// Main kernel converts per-node A operands to bf16 in LDS once, then runs the
// three neighbor projections on v_wmma_f32_16x16x32_bf16 with pure b128 operand
// loads (no per-WMMA conversion VALU).

#define BB      2
#define NN      2000
#define KNBR    32
#define HH      128
#define DD      32          // per-head dim = H / NHEADS
#define EPSF    1e-6f

typedef __attribute__((ext_vector_type(16))) __bf16 v16bf;
typedef __attribute__((ext_vector_type(8)))  float  v8f;

union ABFrag { v16bf v; unsigned short u[16]; uint4 q[2]; };

__device__ __forceinline__ unsigned short f2bf(float f) {
  unsigned int x = __float_as_uint(f);
  x += 0x7FFFu + ((x >> 16) & 1u);          // round-to-nearest-even
  return (unsigned short)(x >> 16);
}

__device__ __forceinline__ float bf2f(unsigned short b) {
  return __uint_as_float(((unsigned int)b) << 16);
}

// Load a 16-bf16 WMMA operand fragment: K-groups [b] and [b+16] (ISA 7.12.2).
__device__ __forceinline__ v16bf load_frag(const unsigned short* p) {
  ABFrag f;
  f.q[0] = *(const uint4*)(p);
  f.q[1] = *(const uint4*)(p + 16);
  return f.v;
}

// Pack a float4 into two u32s holding 4 bf16 (little-endian lane order).
__device__ __forceinline__ uint2 pack4(float4 f) {
  uint2 r;
  r.x = (unsigned int)f2bf(f.x) | ((unsigned int)f2bf(f.y) << 16);
  r.y = (unsigned int)f2bf(f.z) | ((unsigned int)f2bf(f.w) << 16);
  return r;
}

// ---------------- pre-pass: weights fp32 -> bf16 (once per launch) -------------
__global__ __launch_bounds__(256) void cvt_weights(
    const float* __restrict__ wk1, const float* __restrict__ wk2,
    const float* __restrict__ wv,
    unsigned short* __restrict__ o1, unsigned short* __restrict__ o2,
    unsigned short* __restrict__ ov) {
  const int i = blockIdx.x * blockDim.x + threadIdx.x;   // 0 .. 32767
  if (i < HH * HH) { o1[i] = f2bf(wk1[i]); o2[i] = f2bf(wk2[i]); }
  ov[i] = f2bf(wv[i]);                                   // H * 2H = 32768
}

// ---------------- fused main kernel -------------------------------------------
__global__ __launch_bounds__(256) void spatppi_fused(
    const float* __restrict__ h_V,  const float* __restrict__ h_EV,
    const float* __restrict__ h_KV, const float* __restrict__ h_KE,
    const float* __restrict__ mask_V, const float* __restrict__ mask_attend,
    const float* __restrict__ W_Q,  const float* __restrict__ W_O,
    const unsigned short* __restrict__ bK1,   // bf16 W_K1 [128][128]
    const unsigned short* __restrict__ bK2,   // bf16 W_K2 [128][128]
    const unsigned short* __restrict__ bV,    // bf16 W_V  [128][256]
    const float* __restrict__ gain, const float* __restrict__ bias,
    float* __restrict__ out)
{
  __shared__ unsigned short sKE[KNBR * HH];       // 8 KB  bf16 A for K1
  __shared__ unsigned short sKV[KNBR * HH];       // 8 KB  bf16 A for K2
  __shared__ unsigned short sEV[KNBR * 2 * HH];   // 16 KB bf16 A for V
  __shared__ unsigned short sK1[KNBR * HH];       // 8 KB  projected K1 (bf16)
  __shared__ unsigned short sK2[KNBR * HH];       // 8 KB  projected K2 (bf16)
  __shared__ unsigned short sV [KNBR * HH];       // 8 KB  projected V  (bf16)
  __shared__ float sQ[HH];
  __shared__ float sHup[HH];
  __shared__ float sRed[8], sRed2[8];

  const int node = blockIdx.x;        // flattened (b, n)
  const int tid  = threadIdx.x;
  const int wave = tid >> 5;
  const int lane = tid & 31;
  const int ln   = lane & 15;         // row/col inside a 16-wide tile
  const int half = lane >> 4;         // K-range half select (ISA 7.12.2)

  const float* hVrow = h_V + (size_t)node * HH;
  const float* AKE = h_KE + (size_t)node * KNBR * HH;
  const float* AKV = h_KV + (size_t)node * KNBR * HH;
  const float* AEV = h_EV + (size_t)node * KNBR * (2 * HH);

  // ---- stage per-node A operands into LDS as bf16 (coalesced, once) ----
  #pragma unroll
  for (int i = tid * 4; i < KNBR * HH; i += 256 * 4)
    *(uint2*)&sKE[i] = pack4(*(const float4*)(AKE + i));
  #pragma unroll
  for (int i = tid * 4; i < KNBR * HH; i += 256 * 4)
    *(uint2*)&sKV[i] = pack4(*(const float4*)(AKV + i));
  #pragma unroll
  for (int i = tid * 4; i < KNBR * 2 * HH; i += 256 * 4)
    *(uint2*)&sEV[i] = pack4(*(const float4*)(AEV + i));

  // ---- Q = h_V @ W_Q^T (tiny [1,H]x[H,H]; fp32 VALU, waves 0-3) ----
  if (tid < HH) {
    const float* wq = W_Q + (size_t)tid * HH;
    float q = 0.f;
    #pragma unroll 4
    for (int c = 0; c < HH; ++c) q += hVrow[c] * wq[c];
    sQ[tid] = q;
  }
  __syncthreads();

  // ---- K1/K2 projections: 32 tiles (16 each), Kd = 128, 4 unrolled K-steps ----
  for (int t = wave; t < 32; t += 8) {
    const int mat = t >> 4;                 // 0=K1, 1=K2 (wave-uniform)
    const unsigned short* Asrc = mat ? sKV : sKE;
    const unsigned short* Wsrc = mat ? bK2 : bK1;
    unsigned short*       Cdst = mat ? sK2 : sK1;
    const int mt = (t >> 3) & 1;
    const int nt = t & 7;
    const unsigned short* arow = Asrc + (mt * 16 + ln) * HH;
    const unsigned short* wrow = Wsrc + (size_t)(nt * 16 + ln) * HH;

    v8f acc = {};
    #pragma unroll
    for (int ks = 0; ks < 4; ++ks) {
      const int b1 = (ks << 5) + (half << 3);
      acc = __builtin_amdgcn_wmma_f32_16x16x32_bf16(
                false, load_frag(arow + b1), false, load_frag(wrow + b1),
                (short)0, acc, false, false);
    }
    const int row0 = mt * 16 + half * 8;
    const int col  = nt * 16 + ln;
    #pragma unroll
    for (int r = 0; r < 8; ++r) Cdst[(row0 + r) * HH + col] = f2bf(acc[r]);
  }

  // ---- V projection: 16 tiles, Kd = 256, 8 unrolled K-steps ----
  for (int t = wave; t < 16; t += 8) {
    const int mt = t >> 3;
    const int nt = t & 7;
    const unsigned short* arow = sEV + (mt * 16 + ln) * (2 * HH);
    const unsigned short* wrow = bV + (size_t)(nt * 16 + ln) * (2 * HH);

    v8f acc = {};
    #pragma unroll
    for (int ks = 0; ks < 8; ++ks) {
      const int b1 = (ks << 5) + (half << 3);
      acc = __builtin_amdgcn_wmma_f32_16x16x32_bf16(
                false, load_frag(arow + b1), false, load_frag(wrow + b1),
                (short)0, acc, false, false);
    }
    const int row0 = mt * 16 + half * 8;
    const int col  = nt * 16 + ln;
    #pragma unroll
    for (int r = 0; r < 8; ++r) sV[(row0 + r) * HH + col] = f2bf(acc[r]);
  }
  __syncthreads();

  // ---- logits, masked softmax, neighbor aggregation (waves 0-3) ----
  // Thread t < 128: head h = t>>5 (== wave), neighbor k = lane.
  if (tid < HH) {
    const int h = tid >> 5;
    const int k = lane;
    const float* qh = sQ + h * DD;
    const unsigned short* k1r = sK1 + k * HH + h * DD;
    const unsigned short* k2r = sK2 + k * HH + h * DD;
    float lg = 0.f;
    #pragma unroll
    for (int d0 = 0; d0 < DD; d0 += 2) {
      const unsigned int u1 = *(const unsigned int*)(k1r + d0);
      const unsigned int u2 = *(const unsigned int*)(k2r + d0);
      lg += qh[d0]     * __uint_as_float(u1 << 16)
                       * __uint_as_float(u2 << 16);
      lg += qh[d0 + 1] * __uint_as_float(u1 & 0xFFFF0000u)
                       * __uint_as_float(u2 & 0xFFFF0000u);
    }
    lg *= (1.f / (float)DD);
    const float ma = mask_attend[(size_t)node * KNBR + k];
    lg = (ma > 0.f) ? lg : -3.402823466e38f;       // float32 min, as in reference

    float mx = lg;
    #pragma unroll
    for (int off = 16; off > 0; off >>= 1) mx = fmaxf(mx, __shfl_xor(mx, off, 32));
    const float ex = __expf(lg - mx);              // masked lanes -> 0
    float sm = ex;
    #pragma unroll
    for (int off = 16; off > 0; off >>= 1) sm += __shfl_xor(sm, off, 32);
    const float att = ex / sm;

    // h_up[j] = sum_k att[h,k] * V[k,j]; att for (h,k) lives in lane k
    float hup = 0.f;
    #pragma unroll 4
    for (int kk = 0; kk < KNBR; ++kk) {
      const float a = __shfl(att, kk, 32);
      hup += a * bf2f(sV[kk * HH + tid]);
    }
    sHup[tid] = hup;
  }
  __syncthreads();

  // ---- output projection + residual + LayerNorm (unbiased) + node mask ----
  float x = 0.f;
  if (tid < HH) {
    const float* wo = W_O + (size_t)tid * HH;
    float dh = 0.f;
    #pragma unroll 4
    for (int c = 0; c < HH; ++c) dh += sHup[c] * wo[c];
    x = hVrow[tid] + dh;
    float s = x, s2 = x * x;
    #pragma unroll
    for (int off = 16; off > 0; off >>= 1) {
      s  += __shfl_xor(s,  off, 32);
      s2 += __shfl_xor(s2, off, 32);
    }
    if (lane == 0) { sRed[wave] = s; sRed2[wave] = s2; }
  }
  __syncthreads();
  if (tid < HH) {
    const float sum = sRed[0]  + sRed[1]  + sRed[2]  + sRed[3];
    const float sq  = sRed2[0] + sRed2[1] + sRed2[2] + sRed2[3];
    const float mu  = sum * (1.f / (float)HH);
    float var = (sq - (float)HH * mu * mu) * (1.f / (float)(HH - 1));
    var = fmaxf(var, 0.f);
    const float sigma = __fsqrt_rn(var + EPSF);
    const float o = gain[tid] * (x - mu) / (sigma + EPSF) + bias[tid];
    out[(size_t)node * HH + tid] = mask_V[node] * o;
  }
}

extern "C" void kernel_launch(void* const* d_in, const int* in_sizes, int n_in,
                              void* d_out, int out_size, void* d_ws, size_t ws_size,
                              hipStream_t stream) {
  (void)in_sizes; (void)n_in; (void)out_size; (void)ws_size;
  const float* h_V   = (const float*)d_in[0];
  const float* h_EV  = (const float*)d_in[1];
  const float* h_KV  = (const float*)d_in[2];
  const float* h_KE  = (const float*)d_in[3];
  const float* mV    = (const float*)d_in[4];
  const float* mAtt  = (const float*)d_in[5];
  const float* W_Q   = (const float*)d_in[6];
  const float* W_K1  = (const float*)d_in[7];
  const float* W_K2  = (const float*)d_in[8];
  const float* W_V   = (const float*)d_in[9];
  const float* W_O   = (const float*)d_in[10];
  const float* gain  = (const float*)d_in[11];
  const float* bias  = (const float*)d_in[12];
  float* out = (float*)d_out;

  // bf16 weight cache in workspace: [K1: 16384][K2: 16384][V: 32768] ushorts
  unsigned short* bK1 = (unsigned short*)d_ws;
  unsigned short* bK2 = bK1 + HH * HH;
  unsigned short* bK2end = bK2 + HH * HH;
  unsigned short* bWV = bK2end;

  cvt_weights<<<dim3((HH * 2 * HH) / 256), 256, 0, stream>>>(
      W_K1, W_K2, W_V, bK1, bK2, bWV);

  spatppi_fused<<<dim3(BB * NN), 256, 0, stream>>>(
      h_V, h_EV, h_KV, h_KE, mV, mAtt, W_Q, W_O, bK1, bK2, bWV,
      gain, bias, out);
}
